// Encoder_62740882260638
// MI455X (gfx1250) — matrix operand order repaired and checked by reading, not compiled, not run on hardware
//
#include <hip/hip_runtime.h>
#include <stdint.h>

// ---------- WMMA types / helpers (gfx1250, wave32) ----------
typedef __attribute__((ext_vector_type(16))) _Float16 v16h;
typedef __attribute__((ext_vector_type(8)))  float    v8f;
typedef __attribute__((ext_vector_type(4)))  uint32_t v4u;

union Frag16 { v16h v; _Float16 h[16]; uint32_t u[8]; v4u q[2]; };

__device__ __forceinline__ v8f wmma32(v16h a, v16h b, v8f c) {
    // D = A(16x32 f16) * B(32x16 f16) + C(16x16 f32)
    return __builtin_amdgcn_wmma_f32_16x16x32_f16(
        /*neg_a=*/false, a, /*neg_b=*/false, b,
        /*c_mod=*/(short)0, c, /*reuse_a=*/false, /*reuse_b=*/false);
}

// A fragment (16x32) from row-major source. Lane (row, h) needs
// K = kBase+16h..+15 (matches packed-B contiguous order):
// one contiguous 32-byte run -> 2x b128 loads (requires 16B-aligned rows).
__device__ __forceinline__ v16h load_frag_a(const _Float16* A, int lda,
                                            int mBase, int kBase, int lane) {
    const int row  = mBase + (lane & 15);
    const int hsel = lane >> 4;
    const _Float16* Ar = A + (size_t)row * lda + kBase + 16 * hsel;
    Frag16 f;
    f.q[0] = *(const v4u*)(Ar);        // K kBase+16h   .. +7
    f.q[1] = *(const v4u*)(Ar + 8);    // K kBase+16h+8 .. +15
    return f.v;
}

// B fragment (32x16) from fragment-native packed weights:
// packed[((kt*2+h)*N + col)*16 + j] == B_rowmajor[kt*32 + 16h + j][col].
// Lane (col, h) reads 16 contiguous halves -> 2x b128 loads (32B aligned).
__device__ __forceinline__ v16h load_frag_b_packed(const _Float16* Bp, int N,
                                                   int kt, int nBase, int lane) {
    const int col  = nBase + (lane & 15);
    const int hsel = lane >> 4;
    const _Float16* p = Bp + ((size_t)(kt * 2 + hsel) * N + col) * 16;
    Frag16 f;
    f.q[0] = *(const v4u*)(p);
    f.q[1] = *(const v4u*)(p + 8);
    return f.v;
}

// ---------- weight fp32 -> f16 conversion into packed-B layout ----------
// dst linear index i: j=i&15, col=(i>>4)%dstN, kth=(i>>4)/dstN,
// source k = (kth>>1)*32 + (kth&1)*16 + j. Zero-pads beyond srcR/srcC.
__global__ void convert_pack_b_kernel(const float* __restrict__ src,
                                      _Float16* __restrict__ dst,
                                      int srcR, int srcC, int dstK, int dstN) {
    const int total = dstK * dstN;      // dstK multiple of 32
    for (int i = blockIdx.x * blockDim.x + threadIdx.x; i < total;
         i += gridDim.x * blockDim.x) {
        const int j   = i & 15;
        const int col = (i >> 4) % dstN;
        const int kth = (i >> 4) / dstN;
        const int k   = (kth >> 1) * 32 + (kth & 1) * 16 + j;
        float v = (k < srcR && col < srcC) ? src[(size_t)k * srcC + col] : 0.0f;
        dst[i] = (_Float16)v;
    }
}

// ---------- kd branch: input is one-hot(position) -> batch-independent ----------
__global__ void kd_out_kernel(const float* __restrict__ W1, const float* __restrict__ b1,
                              const float* __restrict__ W2, const float* __restrict__ b2,
                              float* __restrict__ kd_out) {
    __shared__ float hid[16 * 136];
    for (int i = threadIdx.x; i < 16 * 136; i += blockDim.x) {
        const int r = i / 136, j = i % 136;
        float v = W1[(size_t)r * 136 + j] + b1[j];
        hid[i] = v > 0.f ? v : 0.f;
    }
    __syncthreads();
    for (int i = threadIdx.x; i < 16 * 256; i += blockDim.x) {
        const int r = i / 256, j = i % 256;
        float s = b2[j];
        for (int k = 0; k < 136; ++k) s += hid[r * 136 + k] * W2[(size_t)k * 256 + j];
        kd_out[i] = s;
    }
}

// ---------- fused per-batch kernel: sort + deepset + main branch ----------
// 1 block per batch, 512 threads = 16 waves (one wave per 16-col output tile).
__global__ __launch_bounds__(512, 1)
void encoder_batch_kernel(
    const float* __restrict__ x, const int* __restrict__ nArr,
    const float* __restrict__ rank_W, const float* __restrict__ rank_b,
    const float* __restrict__ vd_b1, const float* __restrict__ vd_b2,
    const float* __restrict__ ed_W1, const float* __restrict__ ed_b1,
    const float* __restrict__ ed_W2, const float* __restrict__ ed_b2,
    const float* __restrict__ km_W1, const float* __restrict__ km_b1,
    const float* __restrict__ km_b2,
    const float* __restrict__ vm_W1, const float* __restrict__ vm_b1,
    const float* __restrict__ vm_b2,
    const _Float16* __restrict__ vdW1p, const _Float16* __restrict__ vdW2p,
    const _Float16* __restrict__ vmW1ap, const _Float16* __restrict__ vmW2p,
    const _Float16* __restrict__ kmW2p,
    const float* __restrict__ kd_out,
    _Float16* __restrict__ A1em) {

    __shared__ __align__(16) _Float16 s_xs[16][256];   // sorted x, f16
    __shared__ __align__(16) _Float16 s_h1[16][256];   // vd hidden
    __shared__ __align__(16) _Float16 s_hk[16][416];   // km hidden (392 pad 416)
    __shared__ __align__(16) _Float16 s_hv[16][512];   // vm hidden
    __shared__ float s_mag[16];
    __shared__ int   s_perm[16];
    __shared__ float s_y2ds[256];
    __shared__ float s_e1[256];
    __shared__ float s_zds[256];
    __shared__ float s_tvm[512];
    __shared__ float s_tkm[392];
    __shared__ float s_y2[512];

    const int b    = blockIdx.x;
    const int tid  = threadIdx.x;
    const int lane = tid & 31;
    const int wave = tid >> 5;          // 0..15
    const int nb   = nArr[b];

    // --- 1. rank magnitudes: one wave per row, 8 elems/lane + wave reduce ---
    {
        const float* xr = x + ((size_t)b * 16 + wave) * 256;
        float p = 0.f;
        const int k0 = lane * 8;
#pragma unroll
        for (int k = 0; k < 8; ++k) p += xr[k0 + k] * rank_W[k0 + k];
#pragma unroll
        for (int off = 16; off >= 1; off >>= 1) p += __shfl_down(p, off, 32);
        if (lane == 0) {
            float m = p + rank_b[0];
            s_mag[wave] = (wave < nb) ? m : __builtin_inff();  // mask -> +inf
        }
    }
    __syncthreads();

    // --- 2. stable insertion argsort (N=16) by thread 0 ---
    if (tid == 0) {
        float v[16]; int p[16];
        for (int i = 0; i < 16; ++i) { v[i] = s_mag[i]; p[i] = i; }
        for (int i = 1; i < 16; ++i) {
            float fv = v[i]; int fp = p[i]; int j = i - 1;
            while (j >= 0 && v[j] > fv) { v[j + 1] = v[j]; p[j + 1] = p[j]; --j; }
            v[j + 1] = fv; p[j + 1] = fp;
        }
        for (int i = 0; i < 16; ++i) s_perm[i] = p[i];
    }
    __syncthreads();

    // --- 3. gather sorted x into LDS as f16 ---
    for (int i = tid; i < 16 * 256; i += 512) {
        const int r = i >> 8, c = i & 255;
        s_xs[r][c] = (_Float16)x[((size_t)b * 16 + s_perm[r]) * 256 + c];
    }
    __syncthreads();

    // --- 4. H1 = relu(xs @ vd_W1 + b1)  [16x256] via WMMA (1 tile / wave) ---
    {
        v8f acc = {};
        for (int kt = 0; kt < 8; ++kt) {
            v16h a  = load_frag_a(&s_xs[0][0], 256, 0, kt * 32, lane);
            v16h bf = load_frag_b_packed(vdW1p, 256, kt, wave * 16, lane);
            acc = wmma32(a, bf, acc);
        }
        const int col = wave * 16 + (lane & 15);
        const int rb  = (lane >> 4) * 8;
        const float bias = vd_b1[col];
#pragma unroll
        for (int r = 0; r < 8; ++r) {
            float vv = acc[r] + bias;
            s_h1[rb + r][col] = (_Float16)(vv > 0.f ? vv : 0.f);
        }
    }
    __syncthreads();

    // --- 5. vd layer2, gate by kd_out, masked column sum -> y2_ds [256] ---
    {
        v8f acc = {};
        for (int kt = 0; kt < 8; ++kt) {
            v16h a  = load_frag_a(&s_h1[0][0], 256, 0, kt * 32, lane);
            v16h bf = load_frag_b_packed(vdW2p, 256, kt, wave * 16, lane);
            acc = wmma32(a, bf, acc);
        }
        const int col = wave * 16 + (lane & 15);
        const int rb  = (lane >> 4) * 8;
        const float bias = vd_b2[col];
        float part = 0.f;
#pragma unroll
        for (int r = 0; r < 8; ++r) {
            const int row = rb + r;                    // sorted slot; valid = row < nb
            if (row < nb) part += (acc[r] + bias) * kd_out[row * 256 + col];
        }
        const float other = __shfl_down(part, 16, 32); // combine upper half-lanes
        if (lane < 16) s_y2ds[col] = part + other;
    }
    __syncthreads();

    // --- 6. ed MLP (256->256->256) GEMV, fp32 VALU (tiny) ---
    if (tid < 256) {
        float s = ed_b1[tid];
        for (int k = 0; k < 256; ++k) s += s_y2ds[k] * ed_W1[(size_t)k * 256 + tid];
        s_e1[tid] = s > 0.f ? s : 0.f;
    }
    __syncthreads();
    if (tid < 256) {
        float s = ed_b2[tid];
        for (int k = 0; k < 256; ++k) s += s_e1[k] * ed_W2[(size_t)k * 256 + tid];
        s_zds[tid] = s;
    }
    __syncthreads();

    // --- 7. conditioning GEMVs: t_vm = z@vm_W1[256:]+b1, t_km = z@km_W1[17:]+b1 ---
    {
        float s = vm_b1[tid];
        for (int k = 0; k < 256; ++k) s += s_zds[k] * vm_W1[(size_t)(256 + k) * 512 + tid];
        s_tvm[tid] = s;
    }
    if (tid < 392) {
        float s = km_b1[tid];
        for (int k = 0; k < 256; ++k) s += s_zds[k] * km_W1[(size_t)(17 + k) * 392 + tid];
        s_tkm[tid] = s;
    }
    __syncthreads();

    // --- 8. km hidden: one-hot keys select rows of km_W1 (no GEMM needed) ---
    for (int i = tid; i < 16 * 416; i += 512) {
        const int r = i / 416, c = i % 416;
        float vv = 0.f;
        if (c < 392) {
            vv = km_W1[(size_t)r * 392 + c] + s_tkm[c];
            vv = vv > 0.f ? vv : 0.f;
        }
        s_hk[r][c] = (_Float16)vv;
    }

    // --- 9. vm hidden: relu(xs @ vm_W1[:256] + t_vm)  [16x512], 2 tiles / wave ---
    for (int tt = 0; tt < 2; ++tt) {
        const int nt = wave + 16 * tt;
        v8f acc = {};
        for (int kt = 0; kt < 8; ++kt) {
            v16h a  = load_frag_a(&s_xs[0][0], 256, 0, kt * 32, lane);
            v16h bf = load_frag_b_packed(vmW1ap, 512, kt, nt * 16, lane);
            acc = wmma32(a, bf, acc);
        }
        const int col = nt * 16 + (lane & 15);
        const int rb  = (lane >> 4) * 8;
        const float bias = s_tvm[col];
#pragma unroll
        for (int r = 0; r < 8; ++r) {
            float vv = acc[r] + bias;
            s_hv[rb + r][col] = (_Float16)(vv > 0.f ? vv : 0.f);
        }
    }
    __syncthreads();

    // --- 10. y1 = (hv@vm_W2+b)*(hk@km_W2+b), masked column sum -> y2 [512] ---
    for (int tt = 0; tt < 2; ++tt) {
        const int nt = wave + 16 * tt;
        v8f accV = {}, accK = {};
        for (int kt = 0; kt < 16; ++kt) {
            v16h a  = load_frag_a(&s_hv[0][0], 512, 0, kt * 32, lane);
            v16h bf = load_frag_b_packed(vmW2p, 512, kt, nt * 16, lane);
            accV = wmma32(a, bf, accV);
        }
        for (int kt = 0; kt < 13; ++kt) {
            v16h a  = load_frag_a(&s_hk[0][0], 416, 0, kt * 32, lane);
            v16h bf = load_frag_b_packed(kmW2p, 512, kt, nt * 16, lane);
            accK = wmma32(a, bf, accK);
        }
        const int col = nt * 16 + (lane & 15);
        const int rb  = (lane >> 4) * 8;
        const float bV = vm_b2[col], bK = km_b2[col];
        float part = 0.f;
#pragma unroll
        for (int r = 0; r < 8; ++r) {
            const int row = rb + r;
            if (row < nb) part += (accV[r] + bV) * (accK[r] + bK);
        }
        const float other = __shfl_down(part, 16, 32);
        if (lane < 16) s_y2[col] = part + other;
    }
    __syncthreads();

    // --- 11. emit em-GEMM A row: [y2 (512) | one-hot(n) folded into K 512..543] ---
    {
        _Float16* dst = A1em + (size_t)b * 544;
        for (int c = tid; c < 544; c += 512) {
            float vv = (c < 512) ? s_y2[c] : ((c == 512 + nb) ? 1.0f : 0.0f);
            dst[c] = (_Float16)vv;
        }
    }
}

// ---------- batched f16 WMMA GEMM with fused bias/relu (em head) ----------
__global__ __launch_bounds__(256, 1)
void gemm_f16_kernel(const _Float16* __restrict__ A, int lda,
                     const _Float16* __restrict__ Bp, int N,
                     const float* __restrict__ bias, int biasN,
                     _Float16* __restrict__ outH, float* __restrict__ outF, int ldc,
                     int Mtiles, int Ntiles, int Ktiles, int doRelu) {
    const int lane = threadIdx.x & 31;
    const int g = blockIdx.x * 8 + (threadIdx.x >> 5);   // tile id, 8 waves/block
    if (g >= Mtiles * Ntiles) return;
    const int mt = g / Ntiles, nt = g % Ntiles;
    v8f acc = {};
    for (int kt = 0; kt < Ktiles; ++kt) {
        v16h a  = load_frag_a(A, lda, mt * 16, kt * 32, lane);
        v16h bf = load_frag_b_packed(Bp, N, kt, nt * 16, lane);
        acc = wmma32(a, bf, acc);
    }
    const int col = nt * 16 + (lane & 15);
    const int rb  = (lane >> 4) * 8;
    const float bv = (col < biasN) ? bias[col] : 0.f;
#pragma unroll
    for (int r = 0; r < 8; ++r) {
        const int row = mt * 16 + rb + r;
        float vv = acc[r] + bv;
        if (doRelu) vv = vv > 0.f ? vv : 0.f;
        if (outH) outH[(size_t)row * ldc + col] = (_Float16)vv;
        else      outF[(size_t)row * ldc + col] = vv;
    }
}

// ---------- host orchestration ----------
extern "C" void kernel_launch(void* const* d_in, const int* in_sizes, int n_in,
                              void* d_out, int out_size, void* d_ws, size_t ws_size,
                              hipStream_t stream) {
    (void)in_sizes; (void)n_in; (void)out_size; (void)ws_size;
    const float* x      = (const float*)d_in[0];
    const int*   nArr   = (const int*)d_in[1];
    const float* rank_W = (const float*)d_in[2];
    const float* rank_b = (const float*)d_in[3];
    const float* kd_W1 = (const float*)d_in[4];
    const float* kd_b1 = (const float*)d_in[5];
    const float* kd_W2 = (const float*)d_in[6];
    const float* kd_b2 = (const float*)d_in[7];
    const float* vd_W1 = (const float*)d_in[8];
    const float* vd_b1 = (const float*)d_in[9];
    const float* vd_W2 = (const float*)d_in[10];
    const float* vd_b2 = (const float*)d_in[11];
    const float* ed_W1 = (const float*)d_in[12];
    const float* ed_b1 = (const float*)d_in[13];
    const float* ed_W2 = (const float*)d_in[14];
    const float* ed_b2 = (const float*)d_in[15];
    const float* km_W1 = (const float*)d_in[16];
    const float* km_b1 = (const float*)d_in[17];
    const float* km_W2 = (const float*)d_in[18];
    const float* km_b2 = (const float*)d_in[19];
    const float* vm_W1 = (const float*)d_in[20];
    const float* vm_b1 = (const float*)d_in[21];
    const float* vm_W2 = (const float*)d_in[22];
    const float* vm_b2 = (const float*)d_in[23];
    const float* em_W1 = (const float*)d_in[24];
    const float* em_b1 = (const float*)d_in[25];
    const float* em_W2 = (const float*)d_in[26];
    const float* em_b2 = (const float*)d_in[27];

    char* ws = (char*)d_ws;
    size_t off = 0;
    auto alloc = [&](size_t bytes) -> void* {
        void* p = ws + off;
        off += (bytes + 255) & ~(size_t)255;
        return p;
    };
    _Float16* vdW1p  = (_Float16*)alloc((size_t)256 * 256 * 2);
    _Float16* vdW2p  = (_Float16*)alloc((size_t)256 * 256 * 2);
    _Float16* vmW1ap = (_Float16*)alloc((size_t)256 * 512 * 2);
    _Float16* vmW2p  = (_Float16*)alloc((size_t)512 * 512 * 2);
    _Float16* kmW2p  = (_Float16*)alloc((size_t)416 * 512 * 2);
    _Float16* B1em   = (_Float16*)alloc((size_t)544 * 544 * 2);
    _Float16* B2em   = (_Float16*)alloc((size_t)544 * 512 * 2);
    float*    kdOut  = (float*)alloc((size_t)16 * 256 * 4);
    _Float16* A1em   = (_Float16*)alloc((size_t)8192 * 544 * 2);
    _Float16* Hem    = (_Float16*)alloc((size_t)8192 * 544 * 2);

    auto cvt = [&](const float* s, _Float16* d, int sr, int sc, int dK, int dN) {
        int total = dK * dN;
        int blocks = (total + 255) / 256;
        if (blocks > 2048) blocks = 2048;
        convert_pack_b_kernel<<<blocks, 256, 0, stream>>>(s, d, sr, sc, dK, dN);
    };
    cvt(vd_W1, vdW1p, 256, 256, 256, 256);
    cvt(vd_W2, vdW2p, 256, 256, 256, 256);
    cvt(vm_W1, vmW1ap, 256, 512, 256, 512);   // rows 0..255 (x part), contiguous
    cvt(vm_W2, vmW2p, 512, 512, 512, 512);
    cvt(km_W2, kmW2p, 392, 512, 416, 512);    // K padded to 416
    cvt(em_W1, B1em, 529, 520, 544, 544);     // K pad 544 (one-hot rows), N pad 544
    cvt(em_W2, B2em, 520, 512, 544, 512);     // K pad 544

    kd_out_kernel<<<1, 256, 0, stream>>>(kd_W1, kd_b1, kd_W2, kd_b2, kdOut);

    encoder_batch_kernel<<<8192, 512, 0, stream>>>(
        x, nArr, rank_W, rank_b,
        vd_b1, vd_b2,
        ed_W1, ed_b1, ed_W2, ed_b2,
        km_W1, km_b1, km_b2,
        vm_W1, vm_b1, vm_b2,
        vdW1p, vdW2p, vmW1ap, vmW2p, kmW2p,
        kdOut, A1em);

    // em head: H = relu(A1em[8192x544] @ B1em[544x544] + b1)
    {
        const int Mt = 8192 / 16, Nt = 544 / 16, Kt = 544 / 32;
        const int blocks = (Mt * Nt + 7) / 8;
        gemm_f16_kernel<<<blocks, 256, 0, stream>>>(
            A1em, 544, B1em, 544, em_b1, 520, Hem, nullptr, 544, Mt, Nt, Kt, 1);
    }
    // z = H @ B2em[544x512] + b2  -> d_out fp32
    {
        const int Mt = 8192 / 16, Nt = 512 / 16, Kt = 544 / 32;
        const int blocks = (Mt * Nt + 7) / 8;
        gemm_f16_kernel<<<blocks, 256, 0, stream>>>(
            Hem, 544, B2em, 512, em_b2, 512, nullptr, (float*)d_out, 512, Mt, Nt, Kt, 0);
    }
}
